// LMachine_61040075210842
// MI455X (gfx1250) — matrix operand order repaired and checked by reading, not compiled, not run on hardware
//
#include <hip/hip_runtime.h>

// ---------------------------------------------------------------------------
// Plain clang ext-vector types (safe in unions, match gfx1250 WMMA builtins)
// ---------------------------------------------------------------------------
typedef __attribute__((ext_vector_type(16))) __bf16       bf16x16;
typedef __attribute__((ext_vector_type(8)))  float        f32x8;
typedef __attribute__((ext_vector_type(4)))  float        f32x4;
typedef __attribute__((ext_vector_type(4)))  unsigned int u32x4;
typedef __attribute__((ext_vector_type(2)))  unsigned int u32x2;

union Frag {
  bf16x16 v;
  u32x4   q[2];
};

#define KP 40  // LDS row stride in bf16 elements (32 data + 8 pad -> 80B, conflict-free)

static __device__ __forceinline__ unsigned int f2bf(float f) {
  // fp32 -> bf16 bits, round-to-nearest-even
  unsigned int u = __float_as_uint(f);
  unsigned int r = u + 0x7FFFu + ((u >> 16) & 1u);
  return r >> 16;
}
static __device__ __forceinline__ float bf2f(unsigned int b) {
  return __uint_as_float(b << 16);
}

// ---------------------------------------------------------------------------
// C[m,n] = scale(m) * sum_k A[m,k] * Bt[n,k]
//   A  : (M x K) row-major fp32, M % 128 == 0, K % 32 == 0
//   Bt : (N x K) row-major fp32 (i.e. B pre-transposed), N may be ragged
//   fp32 accuracy via bf16 hi/lo split: 3 x v_wmma_f32_16x16x32_bf16 per chunk
//   scale(m) = 1/(rowSum[m]+1e-8) if rowSum != null, else 1
// Block: 256 threads = 8 wave32 waves; tile 128x128x32; wave = 32x64 (2x4 WMMA)
// ---------------------------------------------------------------------------
__global__ __launch_bounds__(256)
void gemm_nt_bf16x3(const float* __restrict__ A,  long long aBatch, int lda,
                    const float* __restrict__ Bt, long long bBatch, int ldb,
                    float* __restrict__ C,        long long cBatch, int ldc,
                    int M, int N, int K,
                    const float* __restrict__ rowSum, long long rsBatch) {
  __shared__ __align__(16) unsigned short sAh[128 * KP];
  __shared__ __align__(16) unsigned short sAl[128 * KP];
  __shared__ __align__(16) unsigned short sBh[128 * KP];
  __shared__ __align__(16) unsigned short sBl[128 * KP];

  const int tid  = threadIdx.x;
  const int lane = tid & 31;
  const int wave = tid >> 5;
  const int bz   = blockIdx.z;

  A  += (long long)bz * aBatch;
  Bt += (long long)bz * bBatch;
  C  += (long long)bz * cBatch;
  if (rowSum) rowSum += (long long)bz * rsBatch;

  const int m0 = blockIdx.y * 128;
  const int n0 = blockIdx.x * 128;

  const int wm = (wave >> 1) * 32;  // wave row offset inside tile (0,32,64,96)
  const int wn = (wave & 1) * 64;   // wave col offset inside tile (0,64)

  const int tr = tid >> 3;          // staging row 0..31 per pass
  const int tc = (tid & 7) << 2;    // staging col (float4) 0..28

  const f32x8 zero = {0.f, 0.f, 0.f, 0.f, 0.f, 0.f, 0.f, 0.f};
  f32x8 acc[2][4];
#pragma unroll
  for (int mt = 0; mt < 2; ++mt)
#pragma unroll
    for (int nt = 0; nt < 4; ++nt) acc[mt][nt] = zero;

  const int frow = lane & 15;  // M-row (A) / N-col (B) owned by this lane
  const int half = lane >> 4;  // lane half selects K interleave group

  for (int k0 = 0; k0 < K; k0 += 32) {
    __syncthreads();  // protect LDS from previous iteration's readers
#pragma unroll
    for (int p = 0; p < 4; ++p) {
      const int row = p * 32 + tr;
      // ---- A tile: always in range (M multiple of 128) ----
      f32x4 fa = *reinterpret_cast<const f32x4*>(A + (long long)(m0 + row) * lda + (k0 + tc));
      {
        unsigned h0 = f2bf(fa.x), h1 = f2bf(fa.y), h2 = f2bf(fa.z), h3 = f2bf(fa.w);
        unsigned l0 = f2bf(fa.x - bf2f(h0)), l1 = f2bf(fa.y - bf2f(h1));
        unsigned l2 = f2bf(fa.z - bf2f(h2)), l3 = f2bf(fa.w - bf2f(h3));
        u32x2 hv = {h0 | (h1 << 16), h2 | (h3 << 16)};
        u32x2 lv = {l0 | (l1 << 16), l2 | (l3 << 16)};
        *reinterpret_cast<u32x2*>(&sAh[row * KP + tc]) = hv;
        *reinterpret_cast<u32x2*>(&sAl[row * KP + tc]) = lv;
      }
      // ---- B tile: N may be ragged (e.g. 2047) -> zero fill ----
      const int gn = n0 + row;
      f32x4 fb = {0.f, 0.f, 0.f, 0.f};
      if (gn < N) fb = *reinterpret_cast<const f32x4*>(Bt + (long long)gn * ldb + (k0 + tc));
      {
        unsigned h0 = f2bf(fb.x), h1 = f2bf(fb.y), h2 = f2bf(fb.z), h3 = f2bf(fb.w);
        unsigned l0 = f2bf(fb.x - bf2f(h0)), l1 = f2bf(fb.y - bf2f(h1));
        unsigned l2 = f2bf(fb.z - bf2f(h2)), l3 = f2bf(fb.w - bf2f(h3));
        u32x2 hv = {h0 | (h1 << 16), h2 | (h3 << 16)};
        u32x2 lv = {l0 | (l1 << 16), l2 | (l3 << 16)};
        *reinterpret_cast<u32x2*>(&sBh[row * KP + tc]) = hv;
        *reinterpret_cast<u32x2*>(&sBl[row * KP + tc]) = lv;
      }
    }
    if (k0 + 32 < K) {  // hint next K-chunk into cache (global_prefetch_b8)
      __builtin_prefetch(A + (long long)(m0 + tr) * lda + (k0 + 32 + tc), 0, 3);
    }
    __syncthreads();

    // ---- A fragments (16-bit A 16x32 layout: lane<16 holds K{0-7,16-23}) ----
    Frag ah[2], al[2];
#pragma unroll
    for (int mt = 0; mt < 2; ++mt) {
      const int off = (wm + mt * 16 + frow) * KP + half * 8;
      ah[mt].q[0] = *reinterpret_cast<const u32x4*>(&sAh[off]);
      ah[mt].q[1] = *reinterpret_cast<const u32x4*>(&sAh[off + 16]);
      al[mt].q[0] = *reinterpret_cast<const u32x4*>(&sAl[off]);
      al[mt].q[1] = *reinterpret_cast<const u32x4*>(&sAl[off + 16]);
    }
    // ---- B fragments (16-bit B 32x16: lane<16 holds K{0-7,16-23}, lane>=16 K{8-15,24-31}) ----
#pragma unroll
    for (int nt = 0; nt < 4; ++nt) {
      const int off = (wn + nt * 16 + frow) * KP + half * 8;
      Frag bh, bl;
      bh.q[0] = *reinterpret_cast<const u32x4*>(&sBh[off]);
      bh.q[1] = *reinterpret_cast<const u32x4*>(&sBh[off + 16]);
      bl.q[0] = *reinterpret_cast<const u32x4*>(&sBl[off]);
      bl.q[1] = *reinterpret_cast<const u32x4*>(&sBl[off + 16]);
#pragma unroll
      for (int mt = 0; mt < 2; ++mt) {
        acc[mt][nt] = __builtin_amdgcn_wmma_f32_16x16x32_bf16(
            false, ah[mt].v, false, bh.v, (short)0, acc[mt][nt], false, false);
        acc[mt][nt] = __builtin_amdgcn_wmma_f32_16x16x32_bf16(
            false, al[mt].v, false, bh.v, (short)0, acc[mt][nt], false, false);
        acc[mt][nt] = __builtin_amdgcn_wmma_f32_16x16x32_bf16(
            false, ah[mt].v, false, bl.v, (short)0, acc[mt][nt], false, false);
      }
    }
  }

  // ---- epilogue: C/D layout -> VGPR r: lane<16 row r, lane>=16 row r+8 ----
  float sc[2][8];
#pragma unroll
  for (int mt = 0; mt < 2; ++mt)
#pragma unroll
    for (int r = 0; r < 8; ++r) {
      if (rowSum) {
        const int grow = m0 + wm + mt * 16 + half * 8 + r;
        sc[mt][r] = 1.f / (rowSum[grow] + 1e-8f);
      } else {
        sc[mt][r] = 1.f;
      }
    }
#pragma unroll
  for (int mt = 0; mt < 2; ++mt) {
    const int rbase = m0 + wm + mt * 16 + half * 8;
#pragma unroll
    for (int nt = 0; nt < 4; ++nt) {
      const int col = n0 + wn + nt * 16 + frow;
      if (col < N) {
#pragma unroll
        for (int r = 0; r < 8; ++r) {
          C[(long long)(rbase + r) * ldc + col] = acc[mt][nt][r] * sc[mt][r];
        }
      }
    }
  }
}

// ---------------------------------------------------------------------------
// Tiled fp32 transpose: in (R x Cc) -> out (Cc x R), batched via grid.z
// ---------------------------------------------------------------------------
__global__ __launch_bounds__(256)
void transpose_k(const float* __restrict__ in, long long inBatch,
                 float* __restrict__ out, long long outBatch, int R, int Cc) {
  __shared__ float tile[32][33];
  const int bz = blockIdx.z;
  in  += (long long)bz * inBatch;
  out += (long long)bz * outBatch;
  const int c0 = blockIdx.x * 32, r0 = blockIdx.y * 32;
  const int tx = threadIdx.x & 31, ty = threadIdx.x >> 5;  // 32 x 8
#pragma unroll
  for (int i = 0; i < 32; i += 8) {
    const int r = r0 + ty + i, c = c0 + tx;
    tile[ty + i][tx] = (r < R && c < Cc) ? in[(long long)r * Cc + c] : 0.f;
  }
  __syncthreads();
#pragma unroll
  for (int i = 0; i < 32; i += 8) {
    const int ro = c0 + ty + i;  // out row = original col
    const int co = r0 + tx;      // out col = original row
    if (ro < Cc && co < R) out[(long long)ro * R + co] = tile[tx][ty + i];
  }
}

// ---------------------------------------------------------------------------
// Diagonal-shear gather + cumsum over l + relu + tril:
//   W[b,l,j] = (j<=l) ? max(0, sum_{m<=l} Agen[b, m, L-1-m+j]) : 0
// One thread per (b, j); reads are coalesced across j at every step l.
// ---------------------------------------------------------------------------
__global__ __launch_bounds__(256)
void scan_kernel(const float* __restrict__ Agen, float* __restrict__ W,
                 int L, int NG) {
  const int j = blockIdx.x * blockDim.x + threadIdx.x;  // 0..L-1
  const int b = blockIdx.y;
  const float* Ab = Agen + (long long)b * L * NG;
  float* Wb = W + (long long)b * L * L;
  float acc = 0.f;
  long long idx = (long long)(L - 1) + j;  // element index of Agen[0, L-1+j]
  for (int l = 0; l < L; ++l) {
    acc += Ab[idx];
    Wb[(long long)l * L + j] = (j <= l) ? fmaxf(acc, 0.f) : 0.f;
    idx += (NG - 1);  // next row, gather column shifts left by one
  }
}

// ---------------------------------------------------------------------------
// rowSum[row] = sum_j W[row, j]   (upper triangle already zeroed)
// ---------------------------------------------------------------------------
__global__ __launch_bounds__(256)
void rowsum_kernel(const float* __restrict__ W, float* __restrict__ rs, int L) {
  const long long row = blockIdx.x;
  const float* p = W + row * (long long)L;
  float s = 0.f;
  for (int j = threadIdx.x; j < L; j += 256) s += p[j];
  __shared__ float red[256];
  red[threadIdx.x] = s;
  __syncthreads();
  for (int st = 128; st > 0; st >>= 1) {
    if (threadIdx.x < st) red[threadIdx.x] += red[threadIdx.x + st];
    __syncthreads();
  }
  if (threadIdx.x == 0) rs[row] = red[0];
}

// ---------------------------------------------------------------------------
extern "C" void kernel_launch(void* const* d_in, const int* in_sizes, int n_in,
                              void* d_out, int out_size, void* d_ws, size_t ws_size,
                              hipStream_t stream) {
  (void)in_sizes; (void)n_in; (void)out_size; (void)ws_size;

  const float* x   = (const float*)d_in[0];  // (B, L, D)
  const float* gen = (const float*)d_in[1];  // (D, 2*MAX_LEN-1) == (512, 2047)
  const float* Wq  = (const float*)d_in[2];  // (D, D)
  const float* Wv  = (const float*)d_in[3];  // (D, D)
  const float* Wc  = (const float*)d_in[4];  // (D, D)
  float* out = (float*)d_out;                // (B, L, D)

  const int Bn = 16, L = 1024, D = 512;
  const int NG = 2 * 1024 - 1;  // 2047; reference slice == full generator here

  // workspace layout (floats); q-region is reused for out2 after stage 2
  float* ws = (float*)d_ws;
  size_t off = 0;
  float* q    = ws + off; off += (size_t)Bn * L * D;   // q, later out2
  float* v    = ws + off; off += (size_t)Bn * L * D;
  float* vT   = ws + off; off += (size_t)Bn * D * L;
  float* genT = ws + off; off += (size_t)NG * D;       // multiple of 4 -> aligned
  float* Agen = ws + off; off += (size_t)Bn * L * NG;
  float* Wm   = ws + off; off += (size_t)Bn * L * L;
  float* rs   = ws + off; off += (size_t)Bn * L;

  const dim3 blk(256);

  // Stage 1: q = x @ Wq^T ; v = x @ Wv^T   (Bt = weight rows directly)
  gemm_nt_bf16x3<<<dim3(D / 128, (Bn * L) / 128, 1), blk, 0, stream>>>(
      x, 0, D, Wq, 0, D, q, 0, D, Bn * L, D, D, nullptr, 0);
  gemm_nt_bf16x3<<<dim3(D / 128, (Bn * L) / 128, 1), blk, 0, stream>>>(
      x, 0, D, Wv, 0, D, v, 0, D, Bn * L, D, D, nullptr, 0);

  // Stage 2 prep: genT = gen^T  (D x NG -> NG x D)
  transpose_k<<<dim3((NG + 31) / 32, (D + 31) / 32, 1), blk, 0, stream>>>(
      gen, 0, genT, 0, D, NG);
  // Stage 2: Agen[b] = q[b] @ gen  (per-batch, ragged N = 2047)
  gemm_nt_bf16x3<<<dim3((NG + 127) / 128, L / 128, Bn), blk, 0, stream>>>(
      q, (long long)L * D, D, genT, 0, D,
      Agen, (long long)L * NG, NG, L, NG, D, nullptr, 0);

  // Stage 3: shear + cumsum + relu + tril, then row sums
  scan_kernel<<<dim3(L / 256, Bn), blk, 0, stream>>>(Agen, Wm, L, NG);
  rowsum_kernel<<<dim3(Bn * L), blk, 0, stream>>>(Wm, rs, L);

  // Stage 4 prep: vT[b] = v[b]^T  (L x D -> D x L)
  transpose_k<<<dim3((D + 31) / 32, (L + 31) / 32, Bn), blk, 0, stream>>>(
      v, (long long)L * D, vT, (long long)D * L, L, D);
  // Stage 4: out2[b] = diag(1/(rowsum+1e-8)) * (W[b] @ v[b])  -> q region
  gemm_nt_bf16x3<<<dim3(D / 128, L / 128, Bn), blk, 0, stream>>>(
      Wm, (long long)L * L, L, vT, (long long)D * L, L,
      q, (long long)L * D, D, L, D, L, rs, (long long)L);

  // Stage 5: out = out2 @ Wc^T
  gemm_nt_bf16x3<<<dim3(D / 128, (Bn * L) / 128, 1), blk, 0, stream>>>(
      q, 0, D, Wc, 0, D, out, 0, D, Bn * L, D, D, nullptr, 0);
}
